// SAGE_82540681494777
// MI455X (gfx1250) — matrix operand order repaired and checked by reading, not compiled, not run on hardware
//
#include <hip/hip_runtime.h>

#define NN 50000
#define NE 800000

typedef __attribute__((ext_vector_type(16))) _Float16 v16h;
typedef __attribute__((ext_vector_type(8)))  float    v8f;

// flat f16 index of A-matrix element (row m, col kc) inside a swizzled
// fragment-major tile: tiles of 32 lanes x 16 halves, one tile per kt.
// From ISA 16-bit A 16x32 layout: kl bits = {v>>2, khalf, v&3, pair}.
__device__ __forceinline__ int a_swz(int m, int kc) {
  int kt    = kc >> 5;
  int kl    = kc & 31;
  int khalf = (kl >> 3) & 1;
  int j     = ((kl >> 4) << 3) | (kl & 7);   // j = 2v+pair, v=((kl>>4)<<2)|((kl>>1)&3)
  int L     = m + (khalf << 4);
  return (kt * 32 + L) * 16 + j;
}

// ---------------------------------------------------------------- zero scratch
__global__ void zero_kernel(float* __restrict__ p, long n) {
  long i = (long)blockIdx.x * blockDim.x + threadIdx.x;
  long stride = (long)gridDim.x * blockDim.x;
  for (; i < n; i += stride) p[i] = 0.0f;
}

// ---------------------------------------------------- layer-1 feature scatter
// A1[dst][0:64) += x[src], A1[dst][64:96) += edge_attr[e], cnt[dst] += 1
__global__ void scatter_edges1(const float* __restrict__ x,
                               const float* __restrict__ ea,
                               const int*   __restrict__ ei,
                               float* __restrict__ A1,
                               float* __restrict__ cnt) {
  unsigned idx = blockIdx.x * 256u + threadIdx.x;
  if (idx >= (unsigned)NE * 96u) return;
  unsigned e = idx / 96u;
  unsigned j = idx - e * 96u;
  int src = ei[e];
  int dst = ei[NE + e];
  float v = (j < 64u) ? x[(long)src * 64 + j] : ea[(long)e * 32 + (j - 64u)];
  unsafeAtomicAdd(&A1[(long)dst * 96 + j], v);
  if (j == 0u) unsafeAtomicAdd(&cnt[dst], 1.0f);
}

// ---------------------------------------------------- layer-2 feature scatter
// Sh[dst][0:128) += h[src]
__global__ void scatter_edges2(const float* __restrict__ h,
                               const int*   __restrict__ ei,
                               float* __restrict__ Sh) {
  unsigned idx = blockIdx.x * 256u + threadIdx.x;
  if (idx >= (unsigned)NE * 128u) return;
  unsigned e = idx >> 7;
  unsigned j = idx & 127u;
  int src = ei[e];
  int dst = ei[NE + e];
  unsafeAtomicAdd(&Sh[(long)dst * 128 + j], h[(long)src * 128 + j]);
}

// ------------------------------------------------------------- fused SAGE layer
// Per 16-node wave tile:
//   aggr = cnt>0 ? (Asum[16xK1] @ Wmsg[K1xN1]) / cnt + bmsg : 0
//   out  = relu( [self(16xDSELF) | aggr(16xN1)] @ Wapp[K2xN2] + bapp )
// All LDS operands held in per-lane WMMA fragment layout -> v16h loads.
template <int K1, int N1, int DSELF, int N2>
__global__ __launch_bounds__(128) void sage_layer(
    const float* __restrict__ a_main, int a_main_w,
    const float* __restrict__ a_ext,  int a_ext_stride,
    const float* __restrict__ selfp,
    const float* __restrict__ cntp,
    const float* __restrict__ Wmsg, const float* __restrict__ bmsg,
    const float* __restrict__ Wapp, const float* __restrict__ bapp,
    float* __restrict__ outp, int nnodes)
{
  constexpr int K2  = DSELF + N1;
  constexpr int NT1 = N1 / 16, KT1 = K1 / 32;
  constexpr int NT2 = N2 / 16, KT2 = K2 / 32;

  // fragment-major LDS: one v16h per (tile, lane)
  __shared__ v16h  sWmsg[KT1 * NT1 * 32];  // B frags, tile = kt*NT1+nt
  __shared__ v16h  sWapp[KT2 * NT2 * 32];
  __shared__ v16h  sA [4][KT1 * 32];       // per-wave A frags (neighbor sums)
  __shared__ v16h  sA2[4][KT2 * 32];       // per-wave A frags ([self | aggr])
  __shared__ float sbmsg[N1];
  __shared__ float sbapp[N2];
  __shared__ float scnt[4][16];

  const int tid   = threadIdx.x;
  const int wave  = tid >> 5;
  const int lane  = tid & 31;
  const int m16   = lane & 15;
  const int khalf = lane >> 4;

  // ---- stage weights into B-fragment layout (f32 -> f16), block-wide.
  // dest flat i: tile=i>>9, L=(i>>4)&31, j=i&15 ; k=kt*32+((j>>3)<<4)+((L>>4)<<3)+(j&7), n=nt*16+(L&15)
  {
    _Float16* w = (_Float16*)sWmsg;
    for (int i = tid; i < K1 * N1; i += 128) {
      int tile = i >> 9, L = (i >> 4) & 31, j = i & 15;
      int kt = tile / NT1, nt = tile - kt * NT1;
      int k = kt * 32 + ((j >> 3) << 4) + ((L >> 4) << 3) + (j & 7);
      int n = nt * 16 + (L & 15);
      w[i] = (_Float16)Wmsg[k * N1 + n];
    }
  }
  {
    _Float16* w = (_Float16*)sWapp;
    for (int i = tid; i < K2 * N2; i += 128) {
      int tile = i >> 9, L = (i >> 4) & 31, j = i & 15;
      int kt = tile / NT2, nt = tile - kt * NT2;
      int k = kt * 32 + ((j >> 3) << 4) + ((L >> 4) << 3) + (j & 7);
      int n = nt * 16 + (L & 15);
      w[i] = (_Float16)Wapp[k * N2 + n];
    }
  }
  for (int i = tid; i < N1; i += 128) sbmsg[i] = bmsg[i];
  for (int i = tid; i < N2; i += 128) sbapp[i] = bapp[i];

  const int base = (blockIdx.x * 4 + wave) * 16;

  // ---- stage neighbor-sum tile [16 x K1] into A-fragment layout.
  // dest flat i: L=(i>>4)&31, j=i&15 ; m=L&15, kc=kt*32+((j>>3)<<4)+((L>>4)<<3)+(j&7)
  {
    _Float16* a = (_Float16*)sA[wave];
    for (int i = lane; i < 16 * K1; i += 32) {
      int kt = i >> 9, L = (i >> 4) & 31, j = i & 15;
      int m  = L & 15;
      int kc = kt * 32 + ((j >> 3) << 4) + ((L >> 4) << 3) + (j & 7);
      int node = base + m;
      float v = 0.0f;
      if (node < nnodes)
        v = (kc < a_main_w) ? a_main[(long)node * a_main_w + kc]
                            : a_ext[(long)node * a_ext_stride + (kc - a_main_w)];
      a[i] = (_Float16)v;
    }
  }
  if (lane < 16) {
    int node = base + lane;
    scnt[wave][lane] = (node < nnodes) ? cntp[node] : 0.0f;
  }
  // ---- stage self features into cols [0, DSELF) of second-GEMM A tile
  {
    _Float16* a = (_Float16*)sA2[wave];
    for (int i = lane; i < 16 * DSELF; i += 32) {
      int r = i / DSELF, c = i - r * DSELF;
      int node = base + r;
      float v = (node < nnodes) ? selfp[(long)node * DSELF + c] : 0.0f;
      a[a_swz(r, c)] = (_Float16)v;
    }
  }
  __syncthreads();

  // ---------------- GEMM1: A[16xK1] @ Wmsg[K1xN1]
  v8f acc[NT1] = {};
#pragma unroll
  for (int kt = 0; kt < KT1; ++kt) {
    v16h afrag = sA[wave][kt * 32 + lane];
#pragma unroll
    for (int nt = 0; nt < NT1; ++nt) {
      v16h bfrag = sWmsg[(kt * NT1 + nt) * 32 + lane];
      acc[nt] = __builtin_amdgcn_wmma_f32_16x16x32_f16(
          false, afrag, false, bfrag, (short)0, acc[nt], false, false);
    }
  }

  // mean + bias (zero for isolated nodes); write aggr into cols [DSELF, K2)
  {
    _Float16* a = (_Float16*)sA2[wave];
#pragma unroll
    for (int nt = 0; nt < NT1; ++nt) {
      int n = nt * 16 + m16;
      float b = sbmsg[n];
#pragma unroll
      for (int r = 0; r < 8; ++r) {
        int m = r + 8 * khalf;
        float c = scnt[wave][m];
        float v = (c > 0.0f) ? (acc[nt][r] / c + b) : 0.0f;
        a[a_swz(m, DSELF + n)] = (_Float16)v;
      }
    }
  }
  __syncthreads();

  // ---------------- GEMM2: A2[16xK2] @ Wapp[K2xN2], relu(+bapp)
  v8f acc2[NT2] = {};
#pragma unroll
  for (int kt = 0; kt < KT2; ++kt) {
    v16h afrag = sA2[wave][kt * 32 + lane];
#pragma unroll
    for (int nt = 0; nt < NT2; ++nt) {
      v16h bfrag = sWapp[(kt * NT2 + nt) * 32 + lane];
      acc2[nt] = __builtin_amdgcn_wmma_f32_16x16x32_f16(
          false, afrag, false, bfrag, (short)0, acc2[nt], false, false);
    }
  }

#pragma unroll
  for (int nt = 0; nt < NT2; ++nt) {
    int n = nt * 16 + m16;
    float b = sbapp[n];
#pragma unroll
    for (int r = 0; r < 8; ++r) {
      int m = r + 8 * khalf;
      int node = base + m;
      if (node < nnodes) {
        float v = acc2[nt][r] + b;
        outp[(long)node * N2 + n] = v > 0.0f ? v : 0.0f;
      }
    }
  }
}

// ----------------------------------------------------------------------- host
extern "C" void kernel_launch(void* const* d_in, const int* in_sizes, int n_in,
                              void* d_out, int out_size, void* d_ws, size_t ws_size,
                              hipStream_t stream) {
  const float* x   = (const float*)d_in[0];
  const int*   ei  = (const int*)d_in[1];     // [2, NE], row0 = src, row1 = dst
  const float* ea  = (const float*)d_in[2];
  const float* Wm1 = (const float*)d_in[3];
  const float* bm1 = (const float*)d_in[4];
  const float* Wa1 = (const float*)d_in[5];
  const float* ba1 = (const float*)d_in[6];
  const float* Wm2 = (const float*)d_in[7];
  const float* bm2 = (const float*)d_in[8];
  const float* Wa2 = (const float*)d_in[9];
  const float* ba2 = (const float*)d_in[10];
  float* out = (float*)d_out;

  float* ws  = (float*)d_ws;
  float* A1  = ws;                       // [NN, 96]  = [sum x[src] | sum ea]
  float* cnt = A1 + (long)NN * 96;       // [NN]
  float* Sh  = cnt + NN;                 // [NN, 128] = sum h[src]
  float* h   = Sh + (long)NN * 128;      // [NN, 128] layer-1 output

  // zero A1 + cnt + Sh (contiguous)
  long zn = (long)NN * 96 + NN + (long)NN * 128;
  zero_kernel<<<1024, 256, 0, stream>>>(ws, zn);

  // layer-1 scatter
  scatter_edges1<<<(NE * 96 + 255) / 256, 256, 0, stream>>>(x, ea, ei, A1, cnt);

  int nb = (NN + 63) / 64;
  // layer-1 dense: K1=96(=64+32), N1=128, DSELF=64, N2=128 -> h
  sage_layer<96, 128, 64, 128><<<nb, 128, 0, stream>>>(
      A1, 96, A1, 96, x, cnt, Wm1, bm1, Wa1, ba1, h, NN);

  // layer-2 scatter of h
  scatter_edges2<<<(NE * 128 + 255) / 256, 256, 0, stream>>>(h, ei, Sh);

  // layer-2 dense: K1=160 (128 from Sh, 32 from A1 cols 64..95), N1=64,
  // DSELF=128, N2=64 -> out
  sage_layer<160, 64, 128, 64><<<nb, 128, 0, stream>>>(
      Sh, 128, A1 + 64, 96, h, cnt, Wm2, bm2, Wa2, ba2, out, NN);
}
